// NeuralMemory_48756468744670
// MI455X (gfx1250) — compile-verified, matrix operand order checked by reading, not run on hardware
//
#include <hip/hip_runtime.h>

// ---------------------------------------------------------------------------
// NeuralMemory (fast-weight linear attention) for gfx1250 / MI455X.
//   k,v,q = x @ W^T + b            (bf16 WMMA, fp32 accum)
//   scan:  out_t = S q_t ; S = d*S + lr*v_t k_t^T   -- chunked form, C=64
//   y     = outs @ Wo^T + bo
// ---------------------------------------------------------------------------

typedef __bf16 bf16;
typedef __attribute__((ext_vector_type(16))) __bf16 v16bf;
typedef __attribute__((ext_vector_type(8)))  __bf16 v8bf;
typedef __attribute__((ext_vector_type(8)))  float  v8f;
typedef __attribute__((ext_vector_type(4)))  float  v4f;

#define DECAY 0.99f
#define LR    0.01f
#define CHUNK 64

#define BB 4
#define SS 4096
#define DD 1024
#define MM 128
#define ROWS (BB*SS)            // 16384

// ---------------- conversions: let clang use native v_cvt ------------------
__device__ __forceinline__ float bf2f(bf16 h) { return (float)h; }
__device__ __forceinline__ bf16  f2bf(float f) { return (bf16)f; }

// ---------------- WMMA fragment loaders (wave32 layouts, CDNA5 ISA) ------
// A-frag: row (lane&15), lane<16 holds K {k0..k0+7, k0+16..k0+23},
//         lane>=16 holds K {k0+8..k0+15, k0+24..k0+31}.
// B-frag is identical with "row" = output column (pass col base as r0).
__device__ __forceinline__ v16bf frag_bf(const bf16* __restrict__ base, int ld,
                                         int r0, int k0, int lane) {
  int r = lane & 15, hi = (lane >> 4) & 1;
  const bf16* p = base + (size_t)(r0 + r) * ld + k0 + hi * 8;
  v8bf lo  = *(const v8bf*)(p);
  v8bf hi8 = *(const v8bf*)(p + 16);
  return __builtin_shufflevector(lo, hi8, 0, 1, 2, 3, 4, 5, 6, 7,
                                 8, 9, 10, 11, 12, 13, 14, 15);
}

// Same fragment but sourced from an fp32 row-major matrix (the LDS state),
// converting to bf16 in-register (native cvt).
__device__ __forceinline__ v16bf frag_f32(const float* __restrict__ base, int ld,
                                          int r0, int k0, int lane) {
  int r = lane & 15, hi = (lane >> 4) & 1;
  const float* p = base + (size_t)(r0 + r) * ld + k0 + hi * 8;
  v4f a0 = *(const v4f*)(p);
  v4f a1 = *(const v4f*)(p + 4);
  v4f b0 = *(const v4f*)(p + 16);
  v4f b1 = *(const v4f*)(p + 20);
  v8bf lo, hi8;
#pragma unroll
  for (int i = 0; i < 4; ++i) {
    lo[i]      = f2bf(a0[i]);
    lo[4 + i]  = f2bf(a1[i]);
    hi8[i]     = f2bf(b0[i]);
    hi8[4 + i] = f2bf(b1[i]);
  }
  return __builtin_shufflevector(lo, hi8, 0, 1, 2, 3, 4, 5, 6, 7,
                                 8, 9, 10, 11, 12, 13, 14, 15);
}

__device__ __forceinline__ v8f wmma_bf(v16bf a, v16bf b, v8f c) {
  return __builtin_amdgcn_wmma_f32_16x16x32_bf16(false, a, false, b,
                                                 (short)0, c, false, false);
}

// ---------------- kernel 1: weights fp32 -> bf16 ----------------------------
__global__ void nm_cvt_weights(const float* __restrict__ Wk,
                               const float* __restrict__ Wv,
                               const float* __restrict__ Wq,
                               const float* __restrict__ Wo,
                               bf16* __restrict__ dst) {
  const int N = MM * DD;  // 131072 per matrix
  for (int i = blockIdx.x * blockDim.x + threadIdx.x; i < N;
       i += gridDim.x * blockDim.x) {
    dst[i]         = f2bf(Wk[i]);
    dst[N + i]     = f2bf(Wv[i]);
    dst[2 * N + i] = f2bf(Wq[i]);
    dst[3 * N + i] = f2bf(Wo[i]);
  }
}

// ---------------- kernel 2: fused k/v/q projection GEMM ---------------------
// One wave computes a 16x128 strip of one projection.  x is converted to bf16
// in-register; W fragments come from the bf16 weight cache (hot in L1/L2).
__global__ void nm_proj(const float* __restrict__ x,
                        const bf16* __restrict__ Wbf,   // Wk,Wv,Wq back-to-back
                        const float* __restrict__ bk,
                        const float* __restrict__ bv,
                        const float* __restrict__ bq,
                        bf16* __restrict__ kvq) {       // k,v,q back-to-back
  int w = threadIdx.x >> 5, lane = threadIdx.x & 31;
  int gw = blockIdx.x * (blockDim.x >> 5) + w;   // 3072 waves total
  int proj  = gw >> 10;                          // 0..2
  int strip = gw & 1023;                         // 0..1023
  const bf16*  W    = Wbf + (size_t)proj * MM * DD;
  const float* bias = (proj == 0) ? bk : (proj == 1) ? bv : bq;
  bf16* out = kvq + (size_t)proj * ROWS * MM;
  int row0 = strip * 16;

  v8f acc[8] = {};
  for (int k0 = 0; k0 < DD; k0 += 32) {
    v16bf a = frag_f32(x, DD, row0, k0, lane);
#pragma unroll
    for (int t = 0; t < 8; ++t) {
      v16bf b = frag_bf(W, DD, t * 16, k0, lane);   // col n = row n of W
      acc[t] = wmma_bf(a, b, acc[t]);
    }
  }
  int n = lane & 15, hi = (lane >> 4) & 1;
#pragma unroll
  for (int t = 0; t < 8; ++t) {
    int col = t * 16 + n;
    float bc = bias[col];
#pragma unroll
    for (int r = 0; r < 8; ++r)
      out[(size_t)(row0 + r + 8 * hi) * MM + col] = f2bf(acc[t][r] + bc);
  }
}

// ---------------- kernel 3: chunked scan (one WG per batch) -----------------
// LDS: Sf   fp32 [128][128]  state                      65536 B
//      Vt   bf16 [128][64]   V^T of current chunk       16384 B
//      Kdt  bf16 [128][64]   (decay-scaled K)^T         16384 B
//      P    bf16 [ 64][64]   LR * tril-decay (Q K^T)     8192 B
//      dpow fp32 [65]        DECAY^i                      260 B
__global__ void nm_scan(const bf16* __restrict__ kvq,
                        bf16* __restrict__ outs,
                        float* __restrict__ state_out) {
  extern __shared__ char smem[];
  float* Sf   = (float*)(smem);
  bf16*  Vt   = (bf16*)(smem + 65536);
  bf16*  Kdt  = (bf16*)(smem + 65536 + 16384);
  bf16*  P    = (bf16*)(smem + 65536 + 32768);
  float* dpow = (float*)(smem + 65536 + 32768 + 8192);

  int b = blockIdx.x;
  int tid = threadIdx.x, w = tid >> 5, lane = tid & 31;
  const bf16* kb = kvq + ((size_t)b * SS) * MM;
  const bf16* vb = kvq + (size_t)ROWS * MM + ((size_t)b * SS) * MM;
  const bf16* qb = kvq + 2 * (size_t)ROWS * MM + ((size_t)b * SS) * MM;

  for (int i = tid; i < MM * MM; i += 256) Sf[i] = 0.0f;
  if (tid <= CHUNK) dpow[tid] = __powf(DECAY, (float)tid);
  __syncthreads();

  for (int ch = 0; ch < SS / CHUNK; ++ch) {
    int t0 = ch * CHUNK;
    const bf16* Kc = kb + (size_t)t0 * MM;
    const bf16* Vc = vb + (size_t)t0 * MM;
    const bf16* Qc = qb + (size_t)t0 * MM;

    // prefetch next chunk's K/V/Q into cache while this chunk computes
    // (global_prefetch_b8; 256 threads x 64B stride covers 16KB per array)
    if (ch + 1 < SS / CHUNK) {
      const bf16* nK = Kc + (size_t)CHUNK * MM;
      const bf16* nV = Vc + (size_t)CHUNK * MM;
      const bf16* nQ = Qc + (size_t)CHUNK * MM;
      __builtin_prefetch(nK + (size_t)tid * 32, 0, 1);
      __builtin_prefetch(nV + (size_t)tid * 32, 0, 1);
      __builtin_prefetch(nQ + (size_t)tid * 32, 0, 1);
    }

    // phase 0: stage V^T and decay-scaled K^T in LDS
    for (int idx = tid; idx < CHUNK * MM; idx += 256) {
      int t = idx >> 7, j = idx & 127;
      Vt[j * CHUNK + t]  = Vc[(size_t)t * MM + j];
      Kdt[j * CHUNK + t] = f2bf(dpow[CHUNK - 1 - t] * bf2f(Kc[(size_t)t * MM + j]));
    }
    __syncthreads();

    // phase 1: P = LR * tril-decay (Q K^T)   (16 tiles, 2 per wave)
#pragma unroll
    for (int c = 0; c < 2; ++c) {
      int tt = w * 2 + c, ti = tt >> 2, tj = tt & 3;
      v8f acc = {};
      for (int k0 = 0; k0 < MM; k0 += 32)
        acc = wmma_bf(frag_bf(Qc, MM, ti * 16, k0, lane),
                      frag_bf(Kc, MM, tj * 16, k0, lane), acc);
      int n = lane & 15, hi = (lane >> 4) & 1;
      int jj = tj * 16 + n;
#pragma unroll
      for (int r = 0; r < 8; ++r) {
        int i = ti * 16 + r + 8 * hi;
        float pv = (jj < i) ? (LR * dpow[i - 1 - jj] * acc[r]) : 0.0f;
        P[i * CHUNK + jj] = f2bf(pv);
      }
    }
    __syncthreads();

    // phase 2: Out = diag(d^i) Q S0^T + P V   (32 tiles, 4 per wave)
#pragma unroll
    for (int c = 0; c < 4; ++c) {
      int tt = w * 4 + c, ti = tt >> 3, tn = tt & 7;
      v8f acc = {};
      for (int k0 = 0; k0 < MM; k0 += 32)
        acc = wmma_bf(frag_bf(Qc, MM, ti * 16, k0, lane),
                      frag_f32(Sf, MM, tn * 16, k0, lane), acc);
      int n = lane & 15, hi = (lane >> 4) & 1;
#pragma unroll
      for (int r = 0; r < 8; ++r) acc[r] *= dpow[ti * 16 + r + 8 * hi];
      for (int k0 = 0; k0 < CHUNK; k0 += 32)
        acc = wmma_bf(frag_bf(P, CHUNK, ti * 16, k0, lane),
                      frag_bf(Vt, CHUNK, tn * 16, k0, lane), acc);
      bf16* og = outs + ((size_t)b * SS + t0) * MM;
#pragma unroll
      for (int r = 0; r < 8; ++r)
        og[(size_t)(ti * 16 + r + 8 * hi) * MM + tn * 16 + n] = f2bf(acc[r]);
    }
    __syncthreads();

    // phase 3: S = d^C * S + LR * (V^T D) K   (64 tiles, wave w owns row tm=w)
    {
      int tm = w;
      float dC = dpow[CHUNK];
      for (int tj = 0; tj < 8; ++tj) {
        v8f acc = {};
        for (int k0 = 0; k0 < CHUNK; k0 += 32)
          acc = wmma_bf(frag_bf(Vt, CHUNK, tm * 16, k0, lane),
                        frag_bf(Kdt, CHUNK, tj * 16, k0, lane), acc);
        int n = lane & 15, hi = (lane >> 4) & 1;
#pragma unroll
        for (int r = 0; r < 8; ++r) {
          int m = tm * 16 + r + 8 * hi, j = tj * 16 + n;
          Sf[m * MM + j] = dC * Sf[m * MM + j] + LR * acc[r];
        }
      }
    }
    __syncthreads();
  }

  // final state -> second output
  for (int i = tid; i < MM * MM; i += 256)
    state_out[(size_t)b * MM * MM + i] = Sf[i];
}

// ---------------- kernel 4: y = outs @ Wo^T + bo ----------------------------
__global__ void nm_outgemm(const bf16* __restrict__ outs,
                           const bf16* __restrict__ Wo_bf,   // [D][M] row-major
                           const float* __restrict__ bo,
                           float* __restrict__ y) {
  int w = threadIdx.x >> 5, lane = threadIdx.x & 31;
  int gw = blockIdx.x * (blockDim.x >> 5) + w;    // 8192 waves
  int strip = gw >> 3, cg = gw & 7;
  int row0 = strip * 16, d0 = cg * 128;

  v8f acc[8] = {};
  for (int k0 = 0; k0 < MM; k0 += 32) {
    v16bf a = frag_bf(outs, MM, row0, k0, lane);
#pragma unroll
    for (int t = 0; t < 8; ++t)
      acc[t] = wmma_bf(a, frag_bf(Wo_bf, MM, d0 + t * 16, k0, lane), acc[t]);
  }
  int n = lane & 15, hi = (lane >> 4) & 1;
#pragma unroll
  for (int t = 0; t < 8; ++t) {
    int d = d0 + t * 16 + n;
    float bd = bo[d];
#pragma unroll
    for (int r = 0; r < 8; ++r)
      y[(size_t)(row0 + r + 8 * hi) * DD + d] = acc[t][r] + bd;
  }
}

// ---------------------------------------------------------------------------
extern "C" void kernel_launch(void* const* d_in, const int* in_sizes, int n_in,
                              void* d_out, int out_size, void* d_ws, size_t ws_size,
                              hipStream_t stream) {
  (void)in_sizes; (void)n_in; (void)out_size; (void)ws_size;
  const float* x  = (const float*)d_in[0];
  const float* Wk = (const float*)d_in[1];
  const float* bk = (const float*)d_in[2];
  const float* Wv = (const float*)d_in[3];
  const float* bv = (const float*)d_in[4];
  const float* Wq = (const float*)d_in[5];
  const float* bq = (const float*)d_in[6];
  const float* Wo = (const float*)d_in[7];
  const float* bo = (const float*)d_in[8];

  // workspace layout (bf16 elements), all 16B aligned
  bf16* ws     = (bf16*)d_ws;
  bf16* Wbf    = ws;                                   // 3 * 131072
  bf16* Wo_bf  = ws + 3 * (size_t)MM * DD;             // 131072
  bf16* kvq    = ws + 4 * (size_t)MM * DD;             // 3 * 16384*128
  bf16* outsb  = kvq + 3 * (size_t)ROWS * MM;          // 16384*128

  float* y         = (float*)d_out;
  float* state_out = y + (size_t)BB * SS * DD;

  nm_cvt_weights<<<256, 256, 0, stream>>>(Wk, Wv, Wq, Wo, Wbf);
  nm_proj<<<384, 256, 0, stream>>>(x, Wbf, bk, bv, bq, kvq);

  size_t smem = 65536 + 16384 + 16384 + 8192 + (CHUNK + 1) * sizeof(float);
  (void)hipFuncSetAttribute((const void*)nm_scan,
                            hipFuncAttributeMaxDynamicSharedMemorySize,
                            (int)smem);
  nm_scan<<<BB, 256, smem, stream>>>(kvq, outsb, state_out);

  nm_outgemm<<<1024, 256, 0, stream>>>(outsb, Wo_bf, bo, y);
}
